// Decoder_69767448756662
// MI455X (gfx1250) — matrix operand order repaired and checked by reading, not compiled, not run on hardware
//
#include <hip/hip_runtime.h>
#include <math.h>

// ---------------- problem constants ----------------
constexpr int T  = 1000;
constexpr int NB = 64;      // batch
constexpr int H  = 1024;    // encoder hidden
constexpr int C  = 256;     // GRU hidden
constexpr int G  = 3 * C;   // 768 gate columns
constexpr long long M = (long long)T * NB;   // 64000 GEMM rows

constexpr size_t ENC_ELEMS = (size_t)T * NB * H;   // 65,536,000
constexpr size_t WIH_ELEMS = (size_t)G * H;        // 786,432
constexpr size_t WHH_ELEMS = (size_t)G * C;        // 196,608

// workspace layout (bytes), 256-aligned
constexpr size_t OFF_ENC = 0;
constexpr size_t OFF_WIH = ((ENC_ELEMS * 2 + 255) / 256) * 256;                 // 131,072,000
constexpr size_t OFF_WHH = OFF_WIH + ((WIH_ELEMS * 2 + 255) / 256) * 256;       // +1,572,864
constexpr size_t OFF_XP  = OFF_WHH + ((WHH_ELEMS * 2 + 255) / 256) * 256;       // +393,216

// scan kernel dynamic LDS: r-gate weights (128KB) + z-gate weights (128KB) + h (32KB)
constexpr size_t LDS_WR_US = (size_t)C * C;        // 65536 ushorts
constexpr size_t LDS_WZ_US = (size_t)C * C;        // 65536 ushorts
constexpr size_t LDS_H_US  = (size_t)NB * C;       // 16384 ushorts
constexpr size_t SCAN_LDS_BYTES = (LDS_WR_US + LDS_WZ_US + LDS_H_US) * 2;  // 294,912 B

// ---------------- WMMA types ----------------
typedef __attribute__((ext_vector_type(16))) __bf16 v16bf;
typedef __attribute__((ext_vector_type(8)))  float  v8f;

union Frag { v16bf v; uint4 q[2]; };

// bf16 helpers (bit-level, round-to-nearest-even)
__device__ __forceinline__ unsigned short f2bf(float f) {
    unsigned int u = __float_as_uint(f);
    u = (u + 0x7FFFu + ((u >> 16) & 1u)) >> 16;
    return (unsigned short)u;
}
__device__ __forceinline__ float bf2f(unsigned short s) {
    return __uint_as_float(((unsigned int)s) << 16);
}

// A fragment (16x32 bf16, M x K): lane m = lane&15 holds row m; 16 contiguous
// K values at k0 + half*16  (half = lane>>4).  Source row-major, K-contig.
__device__ __forceinline__ v16bf load_a_frag(const unsigned short* rowp, int k0, int half) {
    Frag f;
    f.q[0] = *reinterpret_cast<const uint4*>(rowp + k0 + half * 16);
    f.q[1] = *reinterpret_cast<const uint4*>(rowp + k0 + half * 16 + 8);
    return f.v;
}
// B fragment (32x16 bf16, K x N): lane n = lane&15 holds output column n;
// 16 contiguous K values at k0 + half*16.  Source = weight row (out-col) major.
__device__ __forceinline__ v16bf load_b_frag(const unsigned short* rowp, int k0, int half) {
    const uint4* p = reinterpret_cast<const uint4*>(rowp + k0 + half * 16);
    Frag f;
    f.q[0] = p[0];
    f.q[1] = p[1];
    return f.v;
}

__device__ __forceinline__ float sigmoidf(float x) { return 1.0f / (1.0f + expf(-x)); }

// ---------------- kernel 0: fp32 -> bf16 convert ----------------
__global__ void cvt_bf16_kernel(const float* __restrict__ in, unsigned short* __restrict__ out,
                                size_t n) {
    size_t i = (size_t)blockIdx.x * blockDim.x + threadIdx.x;
    size_t stride = (size_t)gridDim.x * blockDim.x;
    for (; i < n; i += stride) out[i] = f2bf(in[i]);
}

// ---------------- kernel 1: xp = enc @ W_ih^T + b_ih ----------------
// A: enc_bf16 [M][1024]   B: W_ih_bf16 [768][1024]   C: xp fp32 [M][768]
// WG tile 128x128, 8 waves, each wave 32x64 (2 m-tiles x 4 n-tiles).
__global__ __launch_bounds__(256) void xp_gemm_kernel(const unsigned short* __restrict__ A,
                                                      const unsigned short* __restrict__ B,
                                                      const float* __restrict__ bias,
                                                      float* __restrict__ Cout) {
    const int mblk = blockIdx.x / 6;       // 500 row blocks
    const int nblk = blockIdx.x % 6;       // 6 col blocks
    const int wave = threadIdx.x >> 5;
    const int lane = threadIdx.x & 31;
    const int n    = lane & 15;
    const int half = lane >> 4;

    const int m0 = mblk * 128 + (wave >> 1) * 32;   // 2 m-tiles start
    const int n0 = nblk * 128 + (wave & 1) * 64;    // 4 n-tiles start

    const unsigned short* arow[2];
    arow[0] = A + (size_t)(m0 + n) * H;
    arow[1] = A + (size_t)(m0 + 16 + n) * H;
    const unsigned short* brow[4];
#pragma unroll
    for (int j = 0; j < 4; ++j) brow[j] = B + (size_t)(n0 + j * 16 + n) * H;

    v8f acc[2][4];
#pragma unroll
    for (int mt = 0; mt < 2; ++mt)
#pragma unroll
        for (int j = 0; j < 4; ++j) acc[mt][j] = (v8f){0,0,0,0,0,0,0,0};

    for (int k0 = 0; k0 < H; k0 += 32) {
        v16bf af[2];
#pragma unroll
        for (int mt = 0; mt < 2; ++mt) af[mt] = load_a_frag(arow[mt], k0, half);
#pragma unroll
        for (int j = 0; j < 4; ++j) {
            v16bf bf = load_b_frag(brow[j], k0, half);
#pragma unroll
            for (int mt = 0; mt < 2; ++mt)
                acc[mt][j] = __builtin_amdgcn_wmma_f32_16x16x32_bf16(
                    false, af[mt], false, bf, (short)0, acc[mt][j], false, false);
        }
    }

#pragma unroll
    for (int mt = 0; mt < 2; ++mt)
#pragma unroll
        for (int j = 0; j < 4; ++j) {
            const int col = n0 + j * 16 + n;
            const float bv = bias[col];
#pragma unroll
            for (int v = 0; v < 8; ++v) {
                const int row = m0 + mt * 16 + v + 8 * half;
                Cout[(size_t)row * G + col] = acc[mt][j][v] + bv;
            }
        }
}

// ---------------- kernel 2: sequential GRU scan ----------------
// Single persistent workgroup (8 wave32), latency-bound serial phase.
// LDS (288 KB dynamic): r-gate + z-gate weights resident (bf16), h single-buffered.
// n-gate weights stream from L2 (128 KB/step).  Per row-tile mt:
//   GEMM phase: all waves read only h rows [16mt,16mt+16) as A-frags (ds_load),
//               6 accumulators (2 col-tiles x r/z/n), one A-frag feeds 6 WMMAs.
//   barrier, then combine phase writes the same rows (owner-lane only) -> no
//   double buffer needed; all concurrent row ranges are disjoint.
__global__ __launch_bounds__(256) void gru_scan_kernel(const float* __restrict__ xp,
                                                       const unsigned short* __restrict__ Whh,
                                                       const float* __restrict__ bhh,
                                                       float* __restrict__ out) {
    extern __shared__ __align__(16) unsigned short smem[];
    unsigned short* wrz = smem;                              // rows 0..511 of W_hh (r+z), 256KB
    unsigned short* hb  = smem + LDS_WR_US + LDS_WZ_US;      // h [64][256] bf16, 32KB

    const int tid  = threadIdx.x;
    const int wave = tid >> 5;
    const int lane = tid & 31;
    const int n    = lane & 15;
    const int half = lane >> 4;

    // stage r/z gate weights into LDS (rows 0..511 are contiguous), zero h
    {
        const uint4* src = reinterpret_cast<const uint4*>(Whh);
        uint4* dst = reinterpret_cast<uint4*>(wrz);
        for (int i = tid; i < (int)((LDS_WR_US + LDS_WZ_US) / 8); i += 256) dst[i] = src[i];
        for (int i = tid; i < NB * C; i += 256) hb[i] = 0;
    }
    __syncthreads();

    const int ct0 = wave * 2;                 // this wave's two h column tiles
    const unsigned short* wr_row[2];
    const unsigned short* wz_row[2];
    const unsigned short* wn_row[2];
#pragma unroll
    for (int q = 0; q < 2; ++q) {
        const int ct = ct0 + q;
        wr_row[q] = wrz + (size_t)(0 * C + ct * 16 + n) * C;          // LDS
        wz_row[q] = wrz + (size_t)(1 * C + ct * 16 + n) * C;          // LDS
        wn_row[q] = Whh + (size_t)(2 * C + ct * 16 + n) * C;          // global (L2)
    }
    float bhr[2], bhz[2], bhn[2];
#pragma unroll
    for (int q = 0; q < 2; ++q) {
        const int c = (ct0 + q) * 16 + n;
        bhr[q] = bhh[c];
        bhz[q] = bhh[C + c];
        bhn[q] = bhh[2 * C + c];
    }

    for (int t = 0; t < T; ++t) {
        const float* xpt = xp + (size_t)t * NB * G;

#pragma unroll
        for (int mt = 0; mt < 4; ++mt) {
            v8f accR[2], accZ[2], accN[2];
#pragma unroll
            for (int q = 0; q < 2; ++q) {
                accR[q] = (v8f){0,0,0,0,0,0,0,0};
                accZ[q] = (v8f){0,0,0,0,0,0,0,0};
                accN[q] = (v8f){0,0,0,0,0,0,0,0};
            }
            const unsigned short* arow = hb + (size_t)(mt * 16 + n) * C;

#pragma unroll
            for (int k8 = 0; k8 < 8; ++k8) {
                const int k0 = k8 * 32;
                v16bf a = load_a_frag(arow, k0, half);
#pragma unroll
                for (int q = 0; q < 2; ++q) {
                    v16bf br = load_b_frag(wr_row[q], k0, half);
                    accR[q] = __builtin_amdgcn_wmma_f32_16x16x32_bf16(
                        false, a, false, br, (short)0, accR[q], false, false);
                    v16bf bz = load_b_frag(wz_row[q], k0, half);
                    accZ[q] = __builtin_amdgcn_wmma_f32_16x16x32_bf16(
                        false, a, false, bz, (short)0, accZ[q], false, false);
                    v16bf bn = load_b_frag(wn_row[q], k0, half);
                    accN[q] = __builtin_amdgcn_wmma_f32_16x16x32_bf16(
                        false, a, false, bn, (short)0, accN[q], false, false);
                }
            }

            __syncthreads();   // all A-frag reads of rows [16mt,16mt+16) done

#pragma unroll
            for (int q = 0; q < 2; ++q) {
                const int c = (ct0 + q) * 16 + n;
#pragma unroll
                for (int v = 0; v < 8; ++v) {
                    const int row = mt * 16 + v + 8 * half;     // batch index
                    const float* xr = xpt + (size_t)row * G;
                    const float r  = sigmoidf(accR[q][v] + bhr[q] + xr[c]);
                    const float z  = sigmoidf(accZ[q][v] + bhz[q] + xr[C + c]);
                    const float hn = accN[q][v] + bhn[q];
                    const float nn = tanhf(xr[2 * C + c] + r * hn);
                    const float hp = bf2f(hb[row * C + c]);
                    const float hnew = (1.0f - z) * nn + z * hp;
                    out[((size_t)t * NB + row) * C + c] = hnew;
                    hb[row * C + c] = f2bf(hnew);
                }
            }
        }
    }
}

// ---------------- kernel 3: in-place log_softmax over C=256 ----------------
__global__ __launch_bounds__(256) void log_softmax_kernel(float* __restrict__ out) {
    const long long row = (long long)blockIdx.x * 8 + (threadIdx.x >> 5);
    const int lane = threadIdx.x & 31;
    float* p = out + (size_t)row * C;

    float x[8];
#pragma unroll
    for (int i = 0; i < 8; ++i) x[i] = p[lane + i * 32];

    float m = x[0];
#pragma unroll
    for (int i = 1; i < 8; ++i) m = fmaxf(m, x[i]);
#pragma unroll
    for (int off = 16; off >= 1; off >>= 1) m = fmaxf(m, __shfl_xor(m, off, 32));

    float s = 0.0f;
#pragma unroll
    for (int i = 0; i < 8; ++i) s += expf(x[i] - m);
#pragma unroll
    for (int off = 16; off >= 1; off >>= 1) s += __shfl_xor(s, off, 32);

    const float ls = logf(s);
#pragma unroll
    for (int i = 0; i < 8; ++i) p[lane + i * 32] = x[i] - m - ls;
}

// ---------------- launch ----------------
extern "C" void kernel_launch(void* const* d_in, const int* in_sizes, int n_in,
                              void* d_out, int out_size, void* d_ws, size_t ws_size,
                              hipStream_t stream) {
    const float* enc  = (const float*)d_in[0];
    const float* W_ih = (const float*)d_in[1];
    const float* W_hh = (const float*)d_in[2];
    const float* b_ih = (const float*)d_in[3];
    const float* b_hh = (const float*)d_in[4];
    float* out = (float*)d_out;

    char* ws = (char*)d_ws;
    unsigned short* enc_bf = (unsigned short*)(ws + OFF_ENC);
    unsigned short* wih_bf = (unsigned short*)(ws + OFF_WIH);
    unsigned short* whh_bf = (unsigned short*)(ws + OFF_WHH);
    float*          xp     = (float*)(ws + OFF_XP);

    // allow 288KB dynamic LDS for the scan kernel (gfx1250 WGP has 320KB)
    hipFuncSetAttribute((const void*)gru_scan_kernel,
                        hipFuncAttributeMaxDynamicSharedMemorySize,
                        (int)SCAN_LDS_BYTES);

    // phase 0: bf16 conversions (bandwidth-bound, one pass each)
    cvt_bf16_kernel<<<4096, 256, 0, stream>>>(enc,  enc_bf, ENC_ELEMS);
    cvt_bf16_kernel<<<1024, 256, 0, stream>>>(W_ih, wih_bf, WIH_ELEMS);
    cvt_bf16_kernel<<<256,  256, 0, stream>>>(W_hh, whh_bf, WHH_ELEMS);

    // phase 1: batched input projection, 64000x1024x768 bf16 WMMA GEMM
    xp_gemm_kernel<<<(64000 / 128) * (G / 128), 256, 0, stream>>>(enc_bf, wih_bf, b_ih, xp);

    // phase 2: sequential recurrence, one persistent workgroup, weights in LDS
    gru_scan_kernel<<<1, 256, SCAN_LDS_BYTES, stream>>>(xp, whh_bf, b_hh, out);

    // phase 3: log_softmax in place on [T*N, 256]
    log_softmax_kernel<<<64000 / 8, 256, 0, stream>>>(out);
}
